// MultiHeadAttention_4389456577415
// MI455X (gfx1250) — compile-verified, matrix operand order checked by reading, not run on hardware
//
#include <hip/hip_runtime.h>
#include <hip/hip_bf16.h>

typedef __attribute__((ext_vector_type(16))) _Float16 v16h;
typedef __attribute__((ext_vector_type(8)))  _Float16 v8h;
typedef __attribute__((ext_vector_type(8)))  float    v8f;
typedef __attribute__((ext_vector_type(4)))  float    v4f;

#define LOG2E 1.44269504088896f

// ---------- fragment chunk loaders (8 contiguous halves = 16B) ----------
__device__ inline v8h ldchunk(const _Float16* p) {
    return *(const v8h*)p;
}
__device__ inline v8h ldchunk(const float* p) {
    v4f a = *(const v4f*)p;
    v4f b = *(const v4f*)(p + 4);
    v8h r;
    r[0] = (_Float16)a[0]; r[1] = (_Float16)a[1];
    r[2] = (_Float16)a[2]; r[3] = (_Float16)a[3];
    r[4] = (_Float16)b[0]; r[5] = (_Float16)b[1];
    r[6] = (_Float16)b[2]; r[7] = (_Float16)b[3];
    return r;
}
__device__ inline v16h mkfrag(v8h lo, v8h hi) {
    v16h r;
#pragma unroll
    for (int i = 0; i < 8; ++i) { r[i] = lo[i]; r[i + 8] = hi[i]; }
    return r;
}
__device__ inline v8f wmma_f16(v16h a, v16h b, v8f c) {
    return __builtin_amdgcn_wmma_f32_16x16x32_f16(false, a, false, b, (short)0, c,
                                                  false, false);
}

// ---------- DPP16 butterfly reductions over each 16-lane half ----------
template <int CTRL>
__device__ inline float dpp16(float x) {
    const int i = __float_as_int(x);
    return __int_as_float(__builtin_amdgcn_update_dpp(i, i, CTRL, 0xF, 0xF, true));
}
__device__ inline float redmax16(float x) {
    x = fmaxf(x, dpp16<0xB1>(x));    // quad_perm xor1
    x = fmaxf(x, dpp16<0x4E>(x));    // quad_perm xor2
    x = fmaxf(x, dpp16<0x141>(x));   // row_half_mirror (^7)
    x = fmaxf(x, dpp16<0x140>(x));   // row_mirror (^15)
    return x;
}
__device__ inline float redsum16(float x) {
    x += dpp16<0xB1>(x);
    x += dpp16<0x4E>(x);
    x += dpp16<0x141>(x);
    x += dpp16<0x140>(x);
    return x;
}

// ---------- CDNA5 async global->LDS (ASYNCcnt-tracked, GV mode) ----------
__device__ inline unsigned ldsoff(const void* p) {
    // generic shared pointer = {SHARED_BASE aperture, 32-bit LDS offset}
    return (unsigned)(unsigned long long)p;
}
__device__ inline void async_ld_b128(unsigned lds_off, const void* gaddr) {
    asm volatile("global_load_async_to_lds_b128 %0, %1, off"
                 :: "v"(lds_off), "v"((unsigned long long)gaddr)
                 : "memory");
}

// =======================================================================
// Projection GEMM: C[M=8192, N=1024] = X @ W^T + bias, K = 1024
// Block: 256 thr = 8 waves, tile 128x128; wave tile 32(M) x 64(N).
// MODE 0: write f16 [B,H,S,64] (Q/K)   MODE 1: write f16 [B,H,64,S] (V^T)
// MODE 2: write f32 row-major [M,1024] (output projection)
// =======================================================================
template <typename TA, int MODE>
__global__ __launch_bounds__(256) void proj_gemm(const TA* __restrict__ X,
                                                 const float* __restrict__ W,
                                                 const float* __restrict__ bias,
                                                 _Float16* __restrict__ dstH,
                                                 float* __restrict__ dstF) {
    const int lane = threadIdx.x & 31;
    const int wave = threadIdx.x >> 5;
    const int wm = wave & 3;          // 4 waves along M
    const int wn = wave >> 2;         // 2 waves along N
    const int r16 = lane & 15;
    const int hi  = lane >> 4;
    const int mBase = blockIdx.x * 128 + wm * 32;
    const int nBase = blockIdx.y * 128 + wn * 64;

    v8f acc[2][4];
#pragma unroll
    for (int i = 0; i < 2; ++i)
#pragma unroll
        for (int j = 0; j < 4; ++j)
#pragma unroll
            for (int e = 0; e < 8; ++e) acc[i][j][e] = 0.f;

    for (int k = 0; k < 1024; k += 32) {
        const int kc = k + hi * 8;      // lane's K-chunk base (ISA A/B 16-bit layout)
        v16h a[2], b[4];
#pragma unroll
        for (int i = 0; i < 2; ++i) {
            const TA* rp = X + (size_t)(mBase + i * 16 + r16) * 1024;
            a[i] = mkfrag(ldchunk(rp + kc), ldchunk(rp + kc + 16));
        }
#pragma unroll
        for (int j = 0; j < 4; ++j) {
            const float* wp = W + (size_t)(nBase + j * 16 + r16) * 1024;
            b[j] = mkfrag(ldchunk(wp + kc), ldchunk(wp + kc + 16));
        }
#pragma unroll
        for (int i = 0; i < 2; ++i)
#pragma unroll
            for (int j = 0; j < 4; ++j)
                acc[i][j] = wmma_f16(a[i], b[j], acc[i][j]);
    }

#pragma unroll
    for (int j = 0; j < 4; ++j) {
        const int col = nBase + j * 16 + r16;
        const float bv = bias[col];
#pragma unroll
        for (int i = 0; i < 2; ++i) {
#pragma unroll
            for (int r = 0; r < 8; ++r) {
                const int m = mBase + i * 16 + r + hi * 8;
                const float v = acc[i][j][r] + bv;
                if (MODE == 2) {
                    dstF[(size_t)m * 1024 + col] = v;
                } else {
                    const int bb = m >> 11, ss = m & 2047;   // batch / seq
                    const int hh = col >> 6, dd = col & 63;  // head / d_k
                    if (MODE == 0)
                        dstH[((size_t)(bb * 16 + hh) * 2048 + ss) * 64 + dd] =
                            (_Float16)v;
                    else
                        dstH[((size_t)(bb * 16 + hh) * 64 + dd) * 2048 + ss] =
                            (_Float16)v;
                }
            }
        }
    }
}

// =======================================================================
// Mask tile flags: flags[qt*64+kc] = any-zero over mask[qt*128..)[kc*32..)
// (mask is [1,1,S,S], broadcast over b,h; one cheap pass, L2-resident)
// =======================================================================
__global__ __launch_bounds__(256) void mask_flags(const int* __restrict__ mask,
                                                  int* __restrict__ flags) {
    const int qt = blockIdx.x >> 6;   // 16 query tiles of 128 rows
    const int kc = blockIdx.x & 63;   // 64 key chunks of 32 cols
    int any0 = 0;
#pragma unroll
    for (int i = 0; i < 16; ++i) {
        const int idx = i * 256 + threadIdx.x;      // 0..4095, coalesced
        const int r = idx >> 5, c = idx & 31;
        any0 |= (mask[(size_t)(qt * 128 + r) * 2048 + kc * 32 + c] == 0);
    }
    any0 = __any(any0);
    __shared__ int s[8];
    if ((threadIdx.x & 31) == 0) s[threadIdx.x >> 5] = any0;
    __syncthreads();
    if (threadIdx.x == 0) {
        int a = 0;
#pragma unroll
        for (int w = 0; w < 8; ++w) a |= s[w];
        flags[blockIdx.x] = a;
    }
}

// =======================================================================
// Flash attention: grid = (S/128, B*H). 8 waves x 16 query rows.
// Qh/Kh: [B,H,S,64] f16.  VhT: [B,H,64,S] f16.  O: [B*S, 1024] f16.
// K/V chunks double-buffered in LDS via async global->LDS prefetch.
// =======================================================================
__global__ __launch_bounds__(256) void flash_fwd(const _Float16* __restrict__ Qh,
                                                 const _Float16* __restrict__ Kh,
                                                 const _Float16* __restrict__ VhT,
                                                 const int* __restrict__ mask,
                                                 const int* __restrict__ flags,
                                                 _Float16* __restrict__ O) {
    __shared__ __align__(16) _Float16 Ks[2][32 * 64];   // [buf][key][d]
    __shared__ __align__(16) _Float16 Vs[2][64 * 32];   // [buf][d][key]
    __shared__ __align__(16) _Float16 Ps[8][16 * 32];   // per-wave P scratch

    const int lane = threadIdx.x & 31;
    const int wave = threadIdx.x >> 5;
    const int r16 = lane & 15;
    const int hi  = lane >> 4;
    const int bh  = blockIdx.y;
    const int qb  = blockIdx.x * 128;
    const int qrow = qb + wave * 16;    // first query row of this wave

    // Q fragments (resident whole kernel), 1/sqrt(d_k)=0.125 folded in (exact in f16)
    v16h qa[2];
    {
        const _Float16* qp = Qh + ((size_t)bh * 2048 + qrow + r16) * 64;
#pragma unroll
        for (int s = 0; s < 2; ++s) {
            const int d0 = s * 32 + hi * 8;
            qa[s] = mkfrag(ldchunk(qp + d0), ldchunk(qp + d0 + 16));
#pragma unroll
            for (int i = 0; i < 16; ++i) qa[s][i] = qa[s][i] * (_Float16)0.125f;
        }
    }

    v8f o[4];
    float mrow[8], lrow[8];
#pragma unroll
    for (int dt = 0; dt < 4; ++dt)
#pragma unroll
        for (int e = 0; e < 8; ++e) o[dt][e] = 0.f;
#pragma unroll
    for (int r = 0; r < 8; ++r) { mrow[r] = -3.0e38f; lrow[r] = 0.f; }

    const int t = threadIdx.x;
    const _Float16* kgp = Kh + (size_t)bh * 2048 * 64;
    const _Float16* vgp = VhT + (size_t)bh * 64 * 2048;
    const int* flagrow = flags + blockIdx.x * 64;

    // this thread's staging slice: one b128 of K, one b128 of V per chunk
    const int krow = t >> 3, kcol = (t & 7) * 8;    // K: 32 rows x 64 halves
    const int vr = t >> 2, vc = (t & 3) * 8;        // V^T: 64 rows x 32 halves

    auto process = [&](int kb, int buf) {
        const _Float16* ks = &Ks[buf][0];
        const _Float16* vs = &Vs[buf][0];

        // scores: S[16 q x 32 keys] = Q * K^T  (load 4 frags, then 4 WMMAs)
        v16h kf[4];
#pragma unroll
        for (int s = 0; s < 2; ++s) {
            const int d0 = s * 32 + hi * 8;
            kf[2 * s] = mkfrag(*(const v8h*)&ks[r16 * 64 + d0],
                               *(const v8h*)&ks[r16 * 64 + d0 + 16]);
            kf[2 * s + 1] = mkfrag(*(const v8h*)&ks[(16 + r16) * 64 + d0],
                                   *(const v8h*)&ks[(16 + r16) * 64 + d0 + 16]);
        }
        v8f c0, c1;
#pragma unroll
        for (int e = 0; e < 8; ++e) { c0[e] = 0.f; c1[e] = 0.f; }
        c0 = wmma_f16(qa[0], kf[0], c0);
        c1 = wmma_f16(qa[0], kf[1], c1);
        c0 = wmma_f16(qa[1], kf[2], c0);
        c1 = wmma_f16(qa[1], kf[3], c1);

        const bool domask = flagrow[kb >> 5] != 0;   // uniform per block

        // online softmax; row = r + 8*hi, 16 cols striped over a 16-lane half
        float p0[8], p1[8];
#pragma unroll
        for (int r = 0; r < 8; ++r) {
            const int qg = qrow + r + hi * 8;
            float s0 = c0[r];    // already scaled via Q
            float s1 = c1[r];
            if (domask) {
                if (mask[(size_t)qg * 2048 + kb + r16] == 0)      s0 = -1e9f;
                if (mask[(size_t)qg * 2048 + kb + 16 + r16] == 0) s1 = -1e9f;
            }
            const float mx = redmax16(fmaxf(s0, s1));
            const float mn = fmaxf(mrow[r], mx);
            const float corr = __builtin_amdgcn_exp2f((mrow[r] - mn) * LOG2E);
            const float e0 = __builtin_amdgcn_exp2f((s0 - mn) * LOG2E);
            const float e1 = __builtin_amdgcn_exp2f((s1 - mn) * LOG2E);
            const float sum = redsum16(e0 + e1);
            lrow[r] = lrow[r] * corr + sum;
            mrow[r] = mn;
            p0[r] = e0; p1[r] = e1;
#pragma unroll
            for (int dt = 0; dt < 4; ++dt) o[dt][r] *= corr;
        }

        // restage P: C-layout -> A-layout via per-wave LDS (in-order per wave)
        _Float16* pw = &Ps[wave][0];
#pragma unroll
        for (int r = 0; r < 8; ++r) {
            pw[(r + 8 * hi) * 32 + r16]      = (_Float16)p0[r];
            pw[(r + 8 * hi) * 32 + 16 + r16] = (_Float16)p1[r];
        }
        const int k0 = hi * 8;
        v16h pf = mkfrag(*(const v8h*)&pw[r16 * 32 + k0],
                         *(const v8h*)&pw[r16 * 32 + k0 + 16]);

        // O += P @ V  (load 4 frags, then 4 WMMAs)
        v16h vf[4];
#pragma unroll
        for (int dt = 0; dt < 4; ++dt)
            vf[dt] = mkfrag(*(const v8h*)&vs[(dt * 16 + r16) * 32 + hi * 8],
                            *(const v8h*)&vs[(dt * 16 + r16) * 32 + hi * 8 + 16]);
#pragma unroll
        for (int dt = 0; dt < 4; ++dt) o[dt] = wmma_f16(pf, vf[dt], o[dt]);
    };

    // prologue: async-load chunk 0 into buffer 0
    async_ld_b128(ldsoff(&Ks[0][krow * 64 + kcol]),
                  kgp + (size_t)krow * 64 + kcol);
    async_ld_b128(ldsoff(&Vs[0][vr * 32 + vc]), vgp + (size_t)vr * 2048 + vc);

    // steady state: unconditional prefetch of next chunk, wait for the two
    // older (current-buffer) async loads -- async loads retire in order.
    int kb = 0;
    for (; kb < 2048 - 32; kb += 32) {
        const int buf = (kb >> 5) & 1;
        const int nb = kb + 32;
        async_ld_b128(ldsoff(&Ks[buf ^ 1][krow * 64 + kcol]),
                      kgp + (size_t)(nb + krow) * 64 + kcol);
        async_ld_b128(ldsoff(&Vs[buf ^ 1][vr * 32 + vc]),
                      vgp + (size_t)vr * 2048 + nb + vc);
        asm volatile("s_wait_asynccnt 0x2" ::: "memory");
        __syncthreads();
        process(kb, buf);
        __syncthreads();
    }
    // peeled last iteration
    asm volatile("s_wait_asynccnt 0x0" ::: "memory");
    __syncthreads();
    process(kb, (kb >> 5) & 1);

    // normalize + write f16 [token, h*64+d]
    const int b = bh >> 4, h = bh & 15;
#pragma unroll
    for (int r = 0; r < 8; ++r) {
        const float inv = lrow[r] > 0.f ? 1.f / lrow[r] : 0.f;
        const size_t tok = (size_t)b * 2048 + qrow + r + 8 * hi;
#pragma unroll
        for (int dt = 0; dt < 4; ++dt)
            O[tok * 1024 + h * 64 + dt * 16 + r16] = (_Float16)(o[dt][r] * inv);
    }
}

// =======================================================================
extern "C" void kernel_launch(void* const* d_in, const int* in_sizes, int n_in,
                              void* d_out, int out_size, void* d_ws, size_t ws_size,
                              hipStream_t stream) {
    const float* q    = (const float*)d_in[0];
    const float* k    = (const float*)d_in[1];
    const float* v    = (const float*)d_in[2];
    const int*   mask = (const int*)d_in[3];
    const float* Wq = (const float*)d_in[4];  const float* bq = (const float*)d_in[5];
    const float* Wk = (const float*)d_in[6];  const float* bk = (const float*)d_in[7];
    const float* Wv = (const float*)d_in[8];  const float* bv = (const float*)d_in[9];
    const float* Wo = (const float*)d_in[10]; const float* bo = (const float*)d_in[11];
    float* out = (float*)d_out;

    const size_t NE = (size_t)8192 * 1024;   // elements per f16 intermediate
    _Float16* Qh  = (_Float16*)d_ws;
    _Float16* Kh  = Qh + NE;
    _Float16* VhT = Kh + NE;
    _Float16* Oa  = VhT + NE;
    int* flags    = (int*)(Oa + NE);         // 1024 ints

    dim3 gG(64, 8), blk(256);
    mask_flags<<<dim3(1024), blk, 0, stream>>>(mask, flags);
    proj_gemm<float, 0><<<gG, blk, 0, stream>>>(q, Wq, bq, Qh, nullptr);
    proj_gemm<float, 0><<<gG, blk, 0, stream>>>(k, Wk, bk, Kh, nullptr);
    proj_gemm<float, 1><<<gG, blk, 0, stream>>>(v, Wv, bv, VhT, nullptr);
    flash_fwd<<<dim3(16, 64), blk, 0, stream>>>(Qh, Kh, VhT, mask, flags, Oa);
    proj_gemm<_Float16, 2><<<gG, blk, 0, stream>>>(Oa, Wo, bo, nullptr, out);
}
